// CrystalGraphConvNet_85143431676002
// MI455X (gfx1250) — matrix operand order repaired and checked by reading, not compiled, not run on hardware
//
#include <hip/hip_runtime.h>
#include <hip/hip_bf16.h>

typedef __attribute__((ext_vector_type(16))) _Float16 v16h;
typedef __attribute__((ext_vector_type(8)))  _Float16 v8h;
typedef __attribute__((ext_vector_type(8)))  float    v8f;

#define N_ATOMS 131072
#define M_NBR   12
#define F_DIM   64
#define NBR_FEA 41
#define KTOT    (2*F_DIM + NBR_FEA)   /* 169 */
#define KPAD    192                   /* 6 chunks of 32 */
#define EMB_K   92
#define EMB_KPAD 96                   /* 3 chunks of 32 */
#define H_DIM   128
#define B_CRY   2048
#define A_CRY   64
#define EPSV    1e-5f

__device__ __forceinline__ float softplusf(float x){
    return fmaxf(x, 0.0f) + log1pf(expf(-fabsf(x)));
}
__device__ __forceinline__ float sigmoidf_(float x){
    return 1.0f / (1.0f + expf(-x));
}

// A-fragment (16-bit A 16x32): lane<16 -> K 0..15, lane>=16 -> K 16..31 (contiguous)
// p must already point at row_base + chunk*32 + hi*16
__device__ __forceinline__ v16h load_fragA(const _Float16* p){
    union { v16h v; v8h h[2]; } u;
    u.h[0] = *(const v8h*)(p);
    u.h[1] = *(const v8h*)(p + 8);
    return u.v;
}
// B-fragment (16-bit B 32x16, same k order as load_fragA): lane<16 -> K 0..15, lane>=16 -> K 16..31 (contiguous)
// p must already point at col_row_base + chunk*32 + hi*16
__device__ __forceinline__ v16h load_fragB(const _Float16* p){
    union { v16h v; v8h h[2]; } u;
    u.h[0] = *(const v8h*)(p);
    u.h[1] = *(const v8h*)(p + 8);
    return u.v;
}

// ---- convert f32 weight [K][Nc] (row-major) to f16 transposed padded [Nc][Kpad] ----
__global__ void wconv_kernel(const float* __restrict__ w, _Float16* __restrict__ out,
                             int K, int Nc, int Kpad){
    int idx = blockIdx.x * blockDim.x + threadIdx.x;
    if (idx < Nc * Kpad){
        int n = idx / Kpad, k = idx % Kpad;
        out[idx] = (_Float16)((k < K) ? w[k * Nc + n] : 0.0f);
    }
}

__global__ void zero_kernel(float* __restrict__ p, int n){
    int i = blockIdx.x * blockDim.x + threadIdx.x;
    if (i < n) p[i] = 0.0f;
}

// ---- embedding: x0 = atom_fea @ emb_w + emb_b via WMMA ----
__global__ __launch_bounds__(256) void embed_kernel(const float* __restrict__ atom_fea,
        const float* __restrict__ emb_b, const _Float16* __restrict__ wT,
        float* __restrict__ x0){
    __shared__ __align__(16) _Float16 tiles[8][16 * EMB_KPAD];
    int wave = threadIdx.x >> 5, lane = threadIdx.x & 31;
    int gw = blockIdx.x * 8 + wave;
    int nw = gridDim.x * 8;
    int r = lane & 15, hi = lane >> 4;
    _Float16* tile = tiles[wave];

    for (int t0 = gw; t0 < N_ATOMS / 16; t0 += nw){
        int abase = t0 * 16;
        for (int row = 0; row < 16; ++row){
            const float* src = atom_fea + (size_t)(abase + row) * EMB_K;
            for (int col = lane; col < EMB_KPAD; col += 32)
                tile[row * EMB_KPAD + col] = (_Float16)((col < EMB_K) ? src[col] : 0.0f);
        }
        asm volatile("s_wait_dscnt 0x0" ::: "memory");
        v16h a0 = load_fragA(tile + r * EMB_KPAD +  0 + hi * 16);
        v16h a1 = load_fragA(tile + r * EMB_KPAD + 32 + hi * 16);
        v16h a2 = load_fragA(tile + r * EMB_KPAD + 64 + hi * 16);
        #pragma unroll
        for (int t = 0; t < 4; ++t){
            float bias = emb_b[t * 16 + r];
            v8f acc;
            #pragma unroll
            for (int i = 0; i < 8; ++i) acc[i] = bias;
            const _Float16* wb = wT + (size_t)(t * 16 + r) * EMB_KPAD + hi * 16;
            v16h b0 = load_fragB(wb +  0);
            v16h b1 = load_fragB(wb + 32);
            v16h b2 = load_fragB(wb + 64);
            acc = __builtin_amdgcn_wmma_f32_16x16x32_f16(false, a0, false, b0, (short)0, acc, false, false);
            acc = __builtin_amdgcn_wmma_f32_16x16x32_f16(false, a1, false, b1, (short)0, acc, false, false);
            acc = __builtin_amdgcn_wmma_f32_16x16x32_f16(false, a2, false, b2, (short)0, acc, false, false);
            #pragma unroll
            for (int i = 0; i < 8; ++i){
                int row = i + hi * 8;                       // C layout: M = vgpr + hi*8, N = r
                x0[(size_t)(abase + row) * F_DIM + t * 16 + r] = acc[i];
            }
        }
    }
}

// ---- conv layer pass kernel (pass 0: bn1 stats, pass 1: apply + summed + bn2 stats) ----
// stats layout: [0..127]=sum1 [128..255]=sumsq1 [256..319]=sum2 [320..383]=sumsq2
// sb1 layout:   [0..127]=scale [128..255]=shift
__global__ __launch_bounds__(256) void conv_pass_kernel(
        const float* __restrict__ x_in, const float* __restrict__ nbr_fea,
        const int* __restrict__ nbr_idx, const _Float16* __restrict__ wT,
        const float* __restrict__ fc_b, float* __restrict__ stats,
        const float* __restrict__ sb1, float* __restrict__ summed, int pass){
    __shared__ __align__(16) _Float16 tiles[8][16 * KPAD];
    int wave = threadIdx.x >> 5, lane = threadIdx.x & 31;
    int gw = blockIdx.x * 8 + wave;
    int nw = gridDim.x * 8;
    int r = lane & 15, hi = lane >> 4;
    _Float16* tile = tiles[wave];

    float sAcc[8], qAcc[8];   // pass 0 per-channel partials (channel = t*16 + r)
    float s2[4],  q2[4];      // pass 1 bn2 partials
    #pragma unroll
    for (int t = 0; t < 8; ++t){ sAcc[t] = 0.f; qAcc[t] = 0.f; }
    #pragma unroll
    for (int t = 0; t < 4; ++t){ s2[t] = 0.f; q2[t] = 0.f; }

    for (int a = gw; a < N_ATOMS; a += nw){
        // build 16x192 A-tile: rows 0..11 = [self | nbr | edge], rows 12..15 = 0, cols 169..191 = 0
        const float* selfp = x_in + (size_t)a * F_DIM;
        for (int row = 0; row < 16; ++row){
            int nbr = (row < M_NBR) ? nbr_idx[a * M_NBR + row] : 0;
            const float* nbrp  = x_in + (size_t)nbr * F_DIM;
            const float* edgep = nbr_fea + ((size_t)a * M_NBR + row) * NBR_FEA;
            for (int col = lane; col < KPAD; col += 32){
                float v = 0.0f;
                if (row < M_NBR){
                    if      (col < F_DIM)     v = selfp[col];
                    else if (col < 2 * F_DIM) v = nbrp[col - F_DIM];
                    else if (col < KTOT)      v = edgep[col - 2 * F_DIM];
                }
                tile[row * KPAD + col] = (_Float16)v;
            }
        }
        asm volatile("s_wait_dscnt 0x0" ::: "memory");
        v16h aF[6];
        #pragma unroll
        for (int c = 0; c < 6; ++c)
            aF[c] = load_fragA(tile + r * KPAD + c * 32 + hi * 16);

        if (pass == 0){
            #pragma unroll
            for (int t = 0; t < 8; ++t){
                float bias = fc_b[t * 16 + r];
                v8f acc;
                #pragma unroll
                for (int i = 0; i < 8; ++i) acc[i] = bias;
                const _Float16* wb = wT + (size_t)(t * 16 + r) * KPAD + hi * 16;
                #pragma unroll
                for (int c = 0; c < 6; ++c){
                    v16h b = load_fragB(wb + c * 32);
                    acc = __builtin_amdgcn_wmma_f32_16x16x32_f16(false, aF[c], false, b, (short)0, acc, false, false);
                }
                // exclude pad rows M=12..15 (high lanes, vgprs 4..7)
                float s = 0.f, q = 0.f;
                #pragma unroll
                for (int i = 0; i < 8; ++i){
                    if (hi == 0 || i < 4){ float v = acc[i]; s += v; q += v * v; }
                }
                sAcc[t] += s; qAcc[t] += q;
            }
        } else {
            #pragma unroll
            for (int t = 0; t < 4; ++t){
                float biasF = fc_b[t * 16 + r];
                float biasC = fc_b[64 + t * 16 + r];
                v8f accF, accC;
                #pragma unroll
                for (int i = 0; i < 8; ++i){ accF[i] = biasF; accC[i] = biasC; }
                const _Float16* wbF = wT + (size_t)(t * 16 + r) * KPAD + hi * 16;
                const _Float16* wbC = wT + (size_t)(64 + t * 16 + r) * KPAD + hi * 16;
                #pragma unroll
                for (int c = 0; c < 6; ++c){
                    v16h bF = load_fragB(wbF + c * 32);
                    accF = __builtin_amdgcn_wmma_f32_16x16x32_f16(false, aF[c], false, bF, (short)0, accF, false, false);
                }
                #pragma unroll
                for (int c = 0; c < 6; ++c){
                    v16h bC = load_fragB(wbC + c * 32);
                    accC = __builtin_amdgcn_wmma_f32_16x16x32_f16(false, aF[c], false, bC, (short)0, accC, false, false);
                }
                float scF = sb1[t * 16 + r],       shF = sb1[128 + t * 16 + r];
                float scC = sb1[64 + t * 16 + r],  shC = sb1[192 + t * 16 + r];
                float s = 0.f;
                #pragma unroll
                for (int i = 0; i < 8; ++i){
                    if (hi == 0 || i < 4){
                        float f  = accF[i] * scF + shF;
                        float cc = accC[i] * scC + shC;
                        s += sigmoidf_(f) * softplusf(cc);
                    }
                }
                float tot = s + __shfl_xor(s, 16, 32);  // combine rows 0..7 (lo) and 8..11 (hi)
                if (!hi) summed[(size_t)a * F_DIM + t * 16 + r] = tot;
                s2[t] += tot; q2[t] += tot * tot;
            }
        }
    }

    if (pass == 0){
        #pragma unroll
        for (int t = 0; t < 8; ++t){
            atomicAdd(&stats[t * 16 + r],       sAcc[t]);
            atomicAdd(&stats[128 + t * 16 + r], qAcc[t]);
        }
    } else if (!hi){
        #pragma unroll
        for (int t = 0; t < 4; ++t){
            atomicAdd(&stats[256 + t * 16 + r], s2[t]);
            atomicAdd(&stats[320 + t * 16 + r], q2[t]);
        }
    }
}

// ---- batchnorm stats -> affine (scale, shift) ----
__global__ void bn_finalize_kernel(const float* __restrict__ stats,
        const float* __restrict__ gamma, const float* __restrict__ beta,
        float* __restrict__ sb, int C, float invCount){
    int i = blockIdx.x * blockDim.x + threadIdx.x;
    if (i < C){
        float mu  = stats[i] * invCount;
        float var = stats[C + i] * invCount - mu * mu;
        float sc  = gamma[i] * rsqrtf(var + EPSV);
        sb[i]     = sc;
        sb[C + i] = beta[i] - mu * sc;
    }
}

// ---- x_new = softplus(x_old + bn2(summed)); summed buffer overwritten in place ----
__global__ void x_update_kernel(const float* __restrict__ x_old,
        float* __restrict__ xs, const float* __restrict__ sb2){
    size_t i = (size_t)blockIdx.x * blockDim.x + threadIdx.x;
    if (i < (size_t)N_ATOMS * F_DIM){
        int ch = (int)(i & 63);
        float s = xs[i] * sb2[ch] + sb2[64 + ch];
        xs[i] = softplusf(x_old[i] + s);
    }
}

// ---- weighted pooling per crystal ----
__global__ __launch_bounds__(64) void pool_kernel(const float* __restrict__ x,
        const float* __restrict__ symm, const int* __restrict__ cidx,
        float* __restrict__ crys){
    __shared__ float wgt[A_CRY];
    __shared__ float tot;
    int b = blockIdx.x, t = threadIdx.x;
    wgt[t] = fabsf(symm[cidx[b * A_CRY + t]]);
    __syncthreads();
    if (t == 0){
        float s = 0.f;
        for (int i = 0; i < A_CRY; ++i) s += wgt[i];
        tot = s;
    }
    __syncthreads();
    float inv = 1.0f / tot;
    float acc = 0.f;
    for (int a = 0; a < A_CRY; ++a){
        int ia = cidx[b * A_CRY + a];
        acc += wgt[a] * inv * x[(size_t)ia * F_DIM + t];
    }
    crys[b * F_DIM + t] = acc;
}

// ---- head: softplus -> fc1 -> softplus -> out ----
__global__ __launch_bounds__(128) void head_kernel(const float* __restrict__ crys,
        const float* __restrict__ fc1_w, const float* __restrict__ fc1_b,
        const float* __restrict__ out_w, const float* __restrict__ out_b,
        float* __restrict__ outp){
    __shared__ float c[F_DIM];
    __shared__ float red[H_DIM];
    int b = blockIdx.x, t = threadIdx.x;
    if (t < F_DIM) c[t] = softplusf(crys[b * F_DIM + t]);
    __syncthreads();
    float h = fc1_b[t];
    for (int f = 0; f < F_DIM; ++f) h += c[f] * fc1_w[f * H_DIM + t];
    h = softplusf(h);
    red[t] = h * out_w[t];
    __syncthreads();
    for (int s = 64; s > 0; s >>= 1){
        if (t < s) red[t] += red[t + s];
        __syncthreads();
    }
    if (t == 0) outp[b] = red[0] + out_b[0];
}

extern "C" void kernel_launch(void* const* d_in, const int* in_sizes, int n_in,
                              void* d_out, int out_size, void* d_ws, size_t ws_size,
                              hipStream_t stream){
    const float* atom_fea  = (const float*)d_in[0];
    const float* atom_symm = (const float*)d_in[1];
    const float* nbr_fea   = (const float*)d_in[2];
    const int*   nbr_idx   = (const int*)d_in[3];
    const int*   cidx      = (const int*)d_in[4];
    const float* emb_w     = (const float*)d_in[5];
    const float* emb_b     = (const float*)d_in[6];
    const float* fc1_w     = (const float*)d_in[25];
    const float* fc1_b     = (const float*)d_in[26];
    const float* out_w     = (const float*)d_in[27];
    const float* out_b     = (const float*)d_in[28];
    float* outp = (float*)d_out;

    char* ws = (char*)d_ws;
    float*    X0    = (float*)(ws);                        // N*64 f32
    float*    X1    = (float*)(ws + 33554432);             // N*64 f32
    _Float16* WT    = (_Float16*)(ws + 67108864);          // up to 128*192 f16
    float*    STATS = (float*)(ws + 67158016);             // 384 f32
    float*    SB1   = (float*)(ws + 67159552);             // 256 f32
    float*    SB2   = (float*)(ws + 67160576);             // 128 f32
    float*    CRYS  = (float*)(ws + 67161088);             // B*64 f32

    // embedding
    wconv_kernel<<<(F_DIM * EMB_KPAD + 255) / 256, 256, 0, stream>>>(emb_w, WT, EMB_K, F_DIM, EMB_KPAD);
    embed_kernel<<<256, 256, 0, stream>>>(atom_fea, emb_b, WT, X0);

    float* xa = X0;
    float* xb = X1;
    for (int L = 0; L < 3; ++L){
        const float* fc_w  = (const float*)d_in[7 + 6 * L + 0];
        const float* fc_b  = (const float*)d_in[7 + 6 * L + 1];
        const float* bn1_g = (const float*)d_in[7 + 6 * L + 2];
        const float* bn1_b = (const float*)d_in[7 + 6 * L + 3];
        const float* bn2_g = (const float*)d_in[7 + 6 * L + 4];
        const float* bn2_b = (const float*)d_in[7 + 6 * L + 5];

        zero_kernel<<<2, 256, 0, stream>>>(STATS, 384);
        wconv_kernel<<<(H_DIM * KPAD + 255) / 256, 256, 0, stream>>>(fc_w, WT, KTOT, H_DIM, KPAD);
        // pass 0: bn1 statistics (recompute GEMM, register-accumulated + few atomics)
        conv_pass_kernel<<<512, 256, 0, stream>>>(xa, nbr_fea, nbr_idx, WT, fc_b, STATS, SB1, xb, 0);
        bn_finalize_kernel<<<1, 128, 0, stream>>>(STATS, bn1_g, bn1_b, SB1, 128,
                                                  1.0f / (float)((size_t)N_ATOMS * M_NBR));
        // pass 1: apply bn1 + gate + neighbor sum -> summed (in xb), bn2 stats
        conv_pass_kernel<<<512, 256, 0, stream>>>(xa, nbr_fea, nbr_idx, WT, fc_b, STATS, SB1, xb, 1);
        bn_finalize_kernel<<<1, 64, 0, stream>>>(STATS + 256, bn2_g, bn2_b, SB2, 64,
                                                 1.0f / (float)N_ATOMS);
        x_update_kernel<<<(N_ATOMS * F_DIM) / 256, 256, 0, stream>>>(xa, xb, SB2);
        float* tmp = xa; xa = xb; xb = tmp;
    }

    pool_kernel<<<B_CRY, 64, 0, stream>>>(xa, atom_symm, cidx, CRYS);
    head_kernel<<<B_CRY, 128, 0, stream>>>(CRYS, fc1_w, fc1_b, out_w, out_b, outp);
}
